// LSTMDetector_21157008900175
// MI455X (gfx1250) — matrix operand order repaired and checked by reading, not compiled, not run on hardware
//
#include <hip/hip_runtime.h>

// CDNA5 / gfx1250 fused LSTM + head, v3.
// grid = B/16 blocks x 256 threads (8 wave32 waves). Each block owns 16 batch
// rows for all T=256 steps. Weights live in per-wave WMMA B-fragments (VGPRs),
// cell state c lives in registers, h ping-pongs through a double-buffered bf16
// A tile in LDS (single barrier per step). Gate columns are permuted per wave
// so the i/f and g/o partners are lane^1 and exchanged with v_mov_b32_dpp8
// (pure VALU, no LDS pipe). Activations use v_rcp_f32 instead of IEEE division,
// and the sigmoid/tanh work is pair-split across partner lanes.

typedef __attribute__((ext_vector_type(16))) __bf16 v16bf;
typedef __attribute__((ext_vector_type(8)))  float  v8f;

#define BSZ   4096
#define TSEQ  256
#define DIN   32
#define HID   64
#define KTOT  96    // D + H
#define MROWS 16

__device__ __forceinline__ float rcp_(float x)      { return __builtin_amdgcn_rcpf(x); }
__device__ __forceinline__ float sigmoidf_(float x) { return rcp_(1.0f + __expf(-x)); }
__device__ __forceinline__ float tanhf_(float x) {
    return fmaf(-2.0f, rcp_(1.0f + __expf(2.0f * x)), 1.0f);   // branch-free
}
__device__ __forceinline__ float dpp8_xor1(float v) {
    // lane selector within groups of 8: {1,0,3,2,5,4,7,6}
    constexpr unsigned imm = 1u | (0u << 3) | (3u << 6) | (2u << 9) |
                             (5u << 12) | (4u << 15) | (7u << 18) | (6u << 21);
    return __int_as_float(__builtin_amdgcn_mov_dpp8(__float_as_int(v), imm));
}

__global__ __launch_bounds__(256)
void lstm_fused_kernel(const float* __restrict__ x,
                       const float* __restrict__ W_ih,
                       const float* __restrict__ W_hh,
                       const float* __restrict__ b_ih,
                       const float* __restrict__ b_hh,
                       const float* __restrict__ W1,
                       const float* __restrict__ b1,
                       const float* __restrict__ W2,
                       const float* __restrict__ b2,
                       float* __restrict__ out)
{
    __shared__ __bf16 Ash[2][MROWS][KTOT];   // ping-pong [x_t | h_{t-1}] (bf16)
    __shared__ float  hsh[MROWS][HID];       // final h_T in f32 (for head)
    __shared__ float  hid1[MROWS][HID / 2];  // head hidden layer

    const int tid   = threadIdx.x;
    const int wav   = tid >> 5;      // 0..7 : wave owns hidden units [8w, 8w+8)
    const int lane  = tid & 31;
    const int lhalf = lane >> 4;     // 0 | 1
    const int l16   = lane & 15;
    const int row0  = blockIdx.x * MROWS;

    // ---- pack B-matrix WMMA fragments (bf16), once -------------------------
    // Column permutation: tile nt, lane l16 -> gate = nt*2 + (l16&1),
    // hidden unit hh = 8*wav + (l16>>1).  Gate order in weights: i,f,g,o.
    // So acc0 holds i (even lanes) / f (odd lanes); acc1 holds g / o.
    // Fragment layout (32x16 KxN, 16-bit): lanes 0-15: N=l16, K=elem 0..15;
    //                                      lanes 16-31: N=l16, K=16+elem.
    v16bf bfr[2][3];
    float biasn[2];
    for (int nt = 0; nt < 2; ++nt) {
        const int gate = nt * 2 + (l16 & 1);
        const int n    = gate * HID + wav * 8 + (l16 >> 1);   // weight row
        biasn[nt] = b_ih[n] + b_hh[n];
        for (int kb = 0; kb < 3; ++kb) {
            const int kbase = kb * 32 + lhalf * 16;
            #pragma unroll
            for (int e = 0; e < 16; ++e) {
                const int K = kbase + e;
                const float v = (K < DIN) ? W_ih[n * DIN + K]
                                          : W_hh[n * HID + (K - DIN)];
                bfr[nt][kb][e] = (__bf16)v;
            }
        }
    }

    // per-lane activation constants for acc1: even lanes need tanh(g) =
    // 2*sig(2x)-1, odd lanes need sig(o) = 1*sig(1*x)+0.
    const bool odd = (l16 & 1) != 0;
    const float k1 = odd ? 1.0f : 2.0f;
    const float k2 = odd ? 1.0f : 2.0f;
    const float k3 = odd ? 0.0f : -1.0f;
    const int  hh  = wav * 8 + (l16 >> 1);   // this lane-pair's hidden unit

    // ---- init: h0 = 0 in A[0]; stage x_0; c in registers = 0 ---------------
    float creg[8];
    #pragma unroll
    for (int r = 0; r < 8; ++r) creg[r] = 0.0f;

    for (int j = tid; j < MROWS * HID; j += 256) {
        const int m = j >> 6, u = j & 63;
        Ash[0][m][DIN + u] = (__bf16)0.0f;
    }
    for (int j = tid; j < MROWS * DIN; j += 256) {
        const int m = j >> 5, d = j & 31;
        Ash[0][m][d] = (__bf16)x[(size_t)(row0 + m) * (TSEQ * DIN) + d]; // t=0
    }
    __syncthreads();

    // ================= recurrent loop: ONE barrier per step =================
    for (int t = 0; t < TSEQ; ++t) {
        const int cur = t & 1, nxt = cur ^ 1;

        // prefetch x_{t+1} into registers; latency hides under the WMMAs
        float xp[2] = {0.0f, 0.0f};
        if (t + 1 < TSEQ) {
            #pragma unroll
            for (int u = 0; u < 2; ++u) {
                const int j = tid + u * 256;
                const int m = j >> 5, d = j & 31;
                xp[u] = x[(size_t)(row0 + m) * (TSEQ * DIN) + (size_t)(t + 1) * DIN + d];
            }
        }

        // load all three A fragments first (distinct regs -> loads pipeline)
        v16bf afr[3];
        #pragma unroll
        for (int kb = 0; kb < 3; ++kb) {
            const int kA = kb * 32 + lhalf * 16;
            #pragma unroll
            for (int e = 0; e < 8; ++e) afr[kb][e]     = Ash[cur][l16][kA + e];
            #pragma unroll
            for (int e = 0; e < 8; ++e) afr[kb][e + 8] = Ash[cur][l16][kA + 8 + e];
        }

        // gates = [x_t | h] @ Wperm^T   (K = 96 = 3 x 32)
        v8f acc0 = {}, acc1 = {};
        #pragma unroll
        for (int kb = 0; kb < 3; ++kb) {
            acc0 = __builtin_amdgcn_wmma_f32_16x16x32_bf16(
                       false, afr[kb], false, bfr[0][kb], (short)0, acc0, false, false);
            acc1 = __builtin_amdgcn_wmma_f32_16x16x32_bf16(
                       false, afr[kb], false, bfr[1][kb], (short)0, acc1, false, false);
        }

        // cell update. acc elem r -> row m = r + 8*lhalf.
        // even lane: acc0=i, acc1=g ; odd lane: acc0=f, acc1=o.
        // pair-split: each lane does 2 activations, dpp8(lane^1) swaps results.
        float hreg[8];
        #pragma unroll
        for (int r = 0; r < 8; ++r) {
            const float g0 = acc0[r] + biasn[0];
            const float g1 = acc1[r] + biasn[1];
            const float a0 = sigmoidf_(g0);                  // sig(i) | sig(f)
            const float a1 = fmaf(k2, sigmoidf_(k1 * g1), k3); // tanh(g) | sig(o)
            const float b0 = dpp8_xor1(a0);                  // sig(f) on even lanes
            const float b1 = dpp8_xor1(a1);                  // sig(o) on even lanes
            const float c  = fmaf(b0, creg[r], a0 * a1);     // f*c + i*tanh(g)
            creg[r] = c;                                     // (garbage on odd lanes)
            hreg[r] = b1 * tanhf_(c);                        // o * tanh(c)
        }
        const bool last = (t == TSEQ - 1);
        if (!odd) {                                          // one writer per element
            #pragma unroll
            for (int r = 0; r < 8; ++r) {
                const int m = r + lhalf * 8;
                Ash[nxt][m][DIN + hh] = (__bf16)hreg[r];
                if (last) hsh[m][hh] = hreg[r];              // f32 copy for the head
            }
        }

        // stage prefetched x_{t+1} into the next buffer
        if (t + 1 < TSEQ) {
            #pragma unroll
            for (int u = 0; u < 2; ++u) {
                const int j = tid + u * 256;
                const int m = j >> 5, d = j & 31;
                Ash[nxt][m][d] = (__bf16)xp[u];
            }
        }
        __syncthreads();
    }

    // ================= head: relu(h W1^T + b1) W2^T + b2 -> sigmoid =========
    for (int p = tid; p < MROWS * (HID / 2); p += 256) {
        const int m = p >> 5, jj = p & 31;
        float s = b1[jj];
        #pragma unroll 8
        for (int k = 0; k < HID; ++k) s += hsh[m][k] * W1[jj * HID + k];
        hid1[m][jj] = fmaxf(s, 0.0f);
    }
    __syncthreads();
    if (tid < MROWS) {
        float s = b2[0];
        #pragma unroll 8
        for (int j = 0; j < HID / 2; ++j) s += hid1[tid][j] * W2[j];
        out[row0 + tid] = sigmoidf_(s);
    }
}

extern "C" void kernel_launch(void* const* d_in, const int* in_sizes, int n_in,
                              void* d_out, int out_size, void* d_ws, size_t ws_size,
                              hipStream_t stream) {
    (void)in_sizes; (void)n_in; (void)out_size; (void)d_ws; (void)ws_size;
    const float* x    = (const float*)d_in[0];
    const float* W_ih = (const float*)d_in[1];
    const float* W_hh = (const float*)d_in[2];
    const float* b_ih = (const float*)d_in[3];
    const float* b_hh = (const float*)d_in[4];
    const float* W1   = (const float*)d_in[5];
    const float* b1   = (const float*)d_in[6];
    const float* W2   = (const float*)d_in[7];
    const float* b2   = (const float*)d_in[8];
    float* out = (float*)d_out;

    dim3 grid(BSZ / MROWS);   // 256 blocks
    dim3 block(256);          // 8 waves (wave32)
    lstm_fused_kernel<<<grid, block, 0, stream>>>(x, W_ih, W_hh, b_ih, b_hh,
                                                  W1, b1, W2, b2, out);
}
